// Frames_patch_65506841199181
// MI455X (gfx1250) — matrix operand order repaired and checked by reading, not compiled, not run on hardware
//
#include <hip/hip_runtime.h>
#include <hip/hip_bf16.h>
#include <math.h>

// ---------------- types ----------------
typedef __attribute__((ext_vector_type(16))) __bf16 v16bf;
typedef __attribute__((ext_vector_type(8)))  float  v8f;
typedef __attribute__((ext_vector_type(4)))  unsigned v4u;

union Frag {
    v16bf v;
    v4u   q[2];
    unsigned u[8];
};

__device__ __forceinline__ unsigned short f2bf(float f) {
    unsigned u = __float_as_uint(f);
    u += 0x7FFFu + ((u >> 16) & 1u);   // round-to-nearest-even
    return (unsigned short)(u >> 16);
}

// =====================================================================
// pack q: x[b,c,h,w] -> q_bf16[b][Q][D]   (patch permutation gather)
// =====================================================================
__global__ __launch_bounds__(256) void pack_q_kernel(
    const float* __restrict__ x, unsigned short* __restrict__ qb,
    int Q, int D, int ph, int pw, int ow)
{
    long long i = (long long)blockIdx.x * 256 + threadIdx.x;   // < B*Q*D
    int dd = (int)(i % D);
    long long r = i / D;
    int qq = (int)(r % Q);
    int bi = (int)(r / Q);
    int pp = ph * pw;
    int ci  = dd / pp;  int rem = dd - ci * pp;
    int phi = rem / pw; int pwi = rem - phi * pw;
    int ohi = qq / ow;  int owi = qq - ohi * ow;
    int y  = ohi * ph + phi;
    int xc = owi * pw + pwi;
    qb[i] = f2bf(x[(((size_t)bi * 256 + ci) * 128 + y) * 128 + xc]);
}

// =====================================================================
// pack k: xs[b*t,c,h,w] -> k_bf16[b][K][D],  K = t*oh*ow (row-major)
// =====================================================================
__global__ __launch_bounds__(256) void pack_k_kernel(
    const float* __restrict__ xs, unsigned short* __restrict__ kb,
    int K, int D, int ph, int pw, int oh, int ow)
{
    long long i = (long long)blockIdx.x * 256 + threadIdx.x;   // < B*K*D
    int dd = (int)(i % D);
    long long r = i / D;
    int kk = (int)(r % K);
    int bi = (int)(r / K);
    int pp = ph * pw;
    int ci  = dd / pp;  int rem = dd - ci * pp;
    int phi = rem / pw; int pwi = rem - phi * pw;
    int QQ = oh * ow;
    int ti = kk / QQ;   int rr = kk - ti * QQ;
    int ohi = rr / ow;  int owi = rr - ohi * ow;
    int y  = ohi * ph + phi;
    int xc = owi * pw + pwi;
    kb[i] = f2bf(xs[(((size_t)(bi * 5 + ti) * 256 + ci) * 128 + y) * 128 + xc]);
}

// =====================================================================
// pack kT: xs -> kT_bf16[b][D][K]  (column-major k; coalesced stores,
// gathered reads of xs are L2-resident). Enables contiguous B-fragment
// loads in GEMM2.
// =====================================================================
__global__ __launch_bounds__(256) void pack_kt_kernel(
    const float* __restrict__ xs, unsigned short* __restrict__ kt,
    int K, int D, int ph, int pw, int oh, int ow)
{
    long long i = (long long)blockIdx.x * 256 + threadIdx.x;   // < B*D*K
    int kk = (int)(i % K);
    long long r = i / K;
    int dd = (int)(r % D);
    int bi = (int)(r / D);
    int pp = ph * pw;
    int ci  = dd / pp;  int rem = dd - ci * pp;
    int phi = rem / pw; int pwi = rem - phi * pw;
    int QQ = oh * ow;
    int ti = kk / QQ;   int rr = kk - ti * QQ;
    int ohi = rr / ow;  int owi = rr - ohi * ow;
    int y  = ohi * ph + phi;
    int xc = owi * pw + pwi;
    kt[i] = f2bf(xs[(((size_t)(bi * 5 + ti) * 256 + ci) * 128 + y) * 128 + xc]);
}

// =====================================================================
// GEMM1: scores[b][Q][K] = q[b] (Q x D) * k[b]^T (D x K)
// one 16x16 tile per wave; A/B frags are contiguous 16B per-lane loads
// =====================================================================
__global__ __launch_bounds__(256) void gemm_qk_kernel(
    const unsigned short* __restrict__ qb, const unsigned short* __restrict__ kb,
    float* __restrict__ sc, int Q, int K, int D)
{
    int wave = blockIdx.x * 8 + (threadIdx.x >> 5);
    int l = threadIdx.x & 31, m = l & 15, g = l >> 4;
    int ktiles = K >> 4, qtiles = Q >> 4;
    int kt = wave % ktiles; wave /= ktiles;
    int qt = wave % qtiles; int bi = wave / qtiles;

    const unsigned short* ar = qb + ((size_t)bi * Q + qt * 16 + m) * D;
    const unsigned short* br = kb + ((size_t)bi * K + kt * 16 + m) * D + 16 * g;

    v8f acc;
#pragma unroll
    for (int r = 0; r < 8; ++r) acc[r] = 0.0f;

    for (int kk = 0; kk < D; kk += 32) {
        Frag a, b;
        a.q[0] = *(const v4u*)(ar + kk + 16 * g);       // K = 16g..16g+7
        a.q[1] = *(const v4u*)(ar + kk + 16 * g + 8);   // K = 16g+8..+15
        b.q[0] = *(const v4u*)(br + kk);                // K = 16g..16g+7
        b.q[1] = *(const v4u*)(br + kk + 8);            // K = 16g+8..+15
        acc = __builtin_amdgcn_wmma_f32_16x16x32_bf16(
            false, a.v, false, b.v, (short)0, acc, false, false);
    }

    float* so = sc + ((size_t)bi * Q + qt * 16) * K + kt * 16 + m;
#pragma unroll
    for (int r = 0; r < 8; ++r)
        so[(size_t)(r + 8 * g) * K] = acc[r];
}

// =====================================================================
// softmax over K with 1/sqrt(d) scale; emits bf16 attention
// =====================================================================
__global__ __launch_bounds__(256) void softmax_kernel(
    const float* __restrict__ sc, unsigned short* __restrict__ at,
    int K, float scale)
{
    __shared__ float red[256];
    int row = blockIdx.x;
    int t = threadIdx.x;
    const float* s = sc + (size_t)row * K;

    float mx = -1e30f;
    for (int i = t; i < K; i += 256) mx = fmaxf(mx, s[i] * scale);
    red[t] = mx; __syncthreads();
    for (int o = 128; o > 0; o >>= 1) { if (t < o) red[t] = fmaxf(red[t], red[t + o]); __syncthreads(); }
    mx = red[0]; __syncthreads();

    float sum = 0.0f;
    for (int i = t; i < K; i += 256) sum += __expf(s[i] * scale - mx);
    red[t] = sum; __syncthreads();
    for (int o = 128; o > 0; o >>= 1) { if (t < o) red[t] += red[t + o]; __syncthreads(); }
    float inv = 1.0f / red[0];

    unsigned short* a = at + (size_t)row * K;
    for (int i = t; i < K; i += 256)
        a[i] = f2bf(__expf(s[i] * scale - mx) * inv);
}

// ---------------------------------------------------------------------
// shared epilogue for GEMM2: scatter D tile into NHWC bf16 concat buf
// ---------------------------------------------------------------------
__device__ __forceinline__ void scatter_val(
    v8f acc, unsigned short* __restrict__ cat, int bi, int qt, int dt,
    int m, int g, int ph, int pw, int ow, int chan_off)
{
    int pp = ph * pw;
    int dd = dt * 16 + m;
    int ci  = dd / pp;  int rem = dd - ci * pp;
    int phi = rem / pw; int pwi = rem - phi * pw;
    unsigned short* cb = cat + (size_t)bi * 128 * 128 * 512;
#pragma unroll
    for (int r = 0; r < 8; ++r) {
        int q = qt * 16 + r + 8 * g;
        int ohi = q / ow, owi = q - ohi * ow;
        int y  = ohi * ph + phi;
        int xc = owi * pw + pwi;
        cb[((size_t)(y * 128 + xc)) * 512 + chan_off + ci] = f2bf(acc[r]);
    }
}

// =====================================================================
// GEMM2 (fast path): val = atten (Q x K) * k (K x D) using kT[b][D][K];
// B fragment = two contiguous 16B loads per lane, same shape as GEMM1.
// =====================================================================
__global__ __launch_bounds__(256) void gemm_av_kt_kernel(
    const unsigned short* __restrict__ at, const unsigned short* __restrict__ kt,
    unsigned short* __restrict__ cat, int Q, int K, int D,
    int ph, int pw, int ow, int chan_off)
{
    int wave = blockIdx.x * 8 + (threadIdx.x >> 5);
    int l = threadIdx.x & 31, m = l & 15, g = l >> 4;
    int dtiles = D >> 4, qtiles = Q >> 4;
    int dt = wave % dtiles; wave /= dtiles;
    int qt = wave % qtiles; int bi = wave / qtiles;

    const unsigned short* ar = at + ((size_t)bi * Q + qt * 16 + m) * K;
    const unsigned short* br = kt + ((size_t)bi * D + dt * 16 + m) * K + 16 * g;

    v8f acc;
#pragma unroll
    for (int r = 0; r < 8; ++r) acc[r] = 0.0f;

    for (int kk = 0; kk < K; kk += 32) {
        Frag a, b;
        a.q[0] = *(const v4u*)(ar + kk + 16 * g);
        a.q[1] = *(const v4u*)(ar + kk + 16 * g + 8);
        b.q[0] = *(const v4u*)(br + kk);
        b.q[1] = *(const v4u*)(br + kk + 8);
        acc = __builtin_amdgcn_wmma_f32_16x16x32_bf16(
            false, a.v, false, b.v, (short)0, acc, false, false);
    }
    scatter_val(acc, cat, bi, qt, dt, m, g, ph, pw, ow, chan_off);
}

// =====================================================================
// GEMM2 (fallback when workspace too small for kT): per-lane u16 gather
// =====================================================================
__global__ __launch_bounds__(256) void gemm_av_kernel(
    const unsigned short* __restrict__ at, const unsigned short* __restrict__ kb,
    unsigned short* __restrict__ cat, int Q, int K, int D,
    int ph, int pw, int ow, int chan_off)
{
    int wave = blockIdx.x * 8 + (threadIdx.x >> 5);
    int l = threadIdx.x & 31, m = l & 15, g = l >> 4;
    int dtiles = D >> 4, qtiles = Q >> 4;
    int dt = wave % dtiles; wave /= dtiles;
    int qt = wave % qtiles; int bi = wave / qtiles;

    const unsigned short* ar = at + ((size_t)bi * Q + qt * 16 + m) * K;
    const unsigned short* bc = kb + (size_t)bi * K * D + dt * 16 + m;

    v8f acc;
#pragma unroll
    for (int r = 0; r < 8; ++r) acc[r] = 0.0f;

    for (int kk = 0; kk < K; kk += 32) {
        Frag a, b;
        a.q[0] = *(const v4u*)(ar + kk + 16 * g);
        a.q[1] = *(const v4u*)(ar + kk + 16 * g + 8);
        const unsigned short* bp = bc + (size_t)(kk + 16 * g) * D;
#pragma unroll
        for (int j = 0; j < 8; ++j) {
            unsigned lo = bp[(size_t)(2 * j) * D];
            unsigned hi = bp[(size_t)(2 * j + 1) * D];
            b.u[j] = lo | (hi << 16);
        }
        acc = __builtin_amdgcn_wmma_f32_16x16x32_bf16(
            false, a.v, false, b.v, (short)0, acc, false, false);
    }
    scatter_val(acc, cat, bi, qt, dt, m, g, ph, pw, ow, chan_off);
}

// =====================================================================
// x -> fused NHWC bf16 channels 0..255
// =====================================================================
__global__ __launch_bounds__(256) void x_to_fused_kernel(
    const float* __restrict__ x, unsigned short* __restrict__ fused)
{
    long long i = (long long)blockIdx.x * 256 + threadIdx.x;  // < B*H*W*256
    int c = (int)(i % 256); long long r = i / 256;
    int xc = (int)(r % 128); r /= 128;
    int y  = (int)(r % 128); int bi = (int)(r / 128);
    fused[((((size_t)bi * 128 + y) * 128 + xc) * 512) + c] =
        f2bf(x[(((size_t)bi * 256 + c) * 128 + y) * 128 + xc]);
}

// =====================================================================
// repack OIHW f32 weights (Cout=256, Cin=512, 3x3) into exact WMMA
// B-fragment layout: wp[nt][ch][lane][j] packed bf16 pairs,
// inner = (ky*3+kx)*512 + ci, 144 chunks of 32
// =====================================================================
__global__ __launch_bounds__(256) void repack_w_kernel(
    const float* __restrict__ w, unsigned* __restrict__ wp)
{
    int i = blockIdx.x * 256 + threadIdx.x;  // < 16*144*32*8
    int j = i & 7; int r = i >> 3;
    int l = r & 31; r >>= 5;
    int ch = r % 144; int nt = r / 144;
    int m = l & 15, g = l >> 4;
    int co = nt * 16 + m;
    unsigned pv = 0;
#pragma unroll
    for (int h = 0; h < 2; ++h) {
        int inner = ch * 32 + 16 * g + 2 * j + h;
        int kyx = inner >> 9; int ci = inner & 511;
        int ky = kyx / 3, kx = kyx - ky * 3;
        pv |= (unsigned)f2bf(w[(((size_t)co * 512 + ci) * 3 + ky) * 3 + kx]) << (16 * h);
    }
    wp[i] = pv;
}

// =====================================================================
// conv_lin: implicit-GEMM 3x3 conv 512->256 over cat NHWC bf16,
// + bias + leaky(0.2), writes fused NHWC channels 256+co
// =====================================================================
__global__ __launch_bounds__(256) void conv_lin_kernel(
    const unsigned short* __restrict__ src, const unsigned* __restrict__ wp,
    const float* __restrict__ bias, unsigned short* __restrict__ dst)
{
    int wave = blockIdx.x * 8 + (threadIdx.x >> 5);
    int l = threadIdx.x & 31, m = l & 15, g = l >> 4;
    int nt = wave & 15; wave >>= 4;
    int pt = wave & 1023; int bi = wave >> 10;
    int pixbase = pt * 16;
    int y  = pixbase >> 7;
    int xb = pixbase & 127;
    const unsigned short* sb = src + (size_t)bi * 128 * 128 * 512;

    v8f acc;
#pragma unroll
    for (int r = 0; r < 8; ++r) acc[r] = 0.0f;

    for (int ch = 0; ch < 144; ++ch) {
        int kyx = ch >> 4; int cb = (ch & 15) * 32;
        int dy = kyx / 3 - 1, dx = kyx - (kyx / 3) * 3 - 1;
        int yy = y + dy, xx = xb + m + dx;
        Frag a, b;
        if (yy >= 0 && yy < 128 && xx >= 0 && xx < 128) {
            const unsigned short* p = sb + ((size_t)(yy * 128 + xx)) * 512 + cb;
            a.q[0] = *(const v4u*)(p + 16 * g);
            a.q[1] = *(const v4u*)(p + 16 * g + 8);
        } else {
#pragma unroll
            for (int j = 0; j < 8; ++j) a.u[j] = 0u;
        }
        const unsigned* bp = wp + ((size_t)(nt * 144 + ch) * 32 + l) * 8;
        b.q[0] = *(const v4u*)bp;
        b.q[1] = *(const v4u*)(bp + 4);
        acc = __builtin_amdgcn_wmma_f32_16x16x32_bf16(
            false, a.v, false, b.v, (short)0, acc, false, false);
    }

    int co = nt * 16 + m;
    float bv = bias[co];
    unsigned short* db = dst + (size_t)bi * 128 * 128 * 512;
#pragma unroll
    for (int r = 0; r < 8; ++r) {
        float v = acc[r] + bv;
        v = v > 0.0f ? v : 0.2f * v;
        int xcol = xb + r + 8 * g;
        db[((size_t)(y * 128 + xcol)) * 512 + 256 + co] = f2bf(v);
    }
}

// =====================================================================
// conv_gatefeat: both 512->256 convs in one wave (shared A fragment),
// fused sigmoid(gate)*leaky(feat) -> d_out NCHW f32
// =====================================================================
__global__ __launch_bounds__(256) void conv_gf_kernel(
    const unsigned short* __restrict__ src,
    const unsigned* __restrict__ wpg, const unsigned* __restrict__ wpf,
    const float* __restrict__ bg, const float* __restrict__ bf_,
    float* __restrict__ out)
{
    int wave = blockIdx.x * 8 + (threadIdx.x >> 5);
    int l = threadIdx.x & 31, m = l & 15, g = l >> 4;
    int nt = wave & 15; wave >>= 4;
    int pt = wave & 1023; int bi = wave >> 10;
    int pixbase = pt * 16;
    int y  = pixbase >> 7;
    int xb = pixbase & 127;
    const unsigned short* sb = src + (size_t)bi * 128 * 128 * 512;

    v8f accg, accf;
#pragma unroll
    for (int r = 0; r < 8; ++r) { accg[r] = 0.0f; accf[r] = 0.0f; }

    for (int ch = 0; ch < 144; ++ch) {
        int kyx = ch >> 4; int cb = (ch & 15) * 32;
        int dy = kyx / 3 - 1, dx = kyx - (kyx / 3) * 3 - 1;
        int yy = y + dy, xx = xb + m + dx;
        Frag a, b0, b1;
        if (yy >= 0 && yy < 128 && xx >= 0 && xx < 128) {
            const unsigned short* p = sb + ((size_t)(yy * 128 + xx)) * 512 + cb;
            a.q[0] = *(const v4u*)(p + 16 * g);
            a.q[1] = *(const v4u*)(p + 16 * g + 8);
        } else {
#pragma unroll
            for (int j = 0; j < 8; ++j) a.u[j] = 0u;
        }
        size_t wo = ((size_t)(nt * 144 + ch) * 32 + l) * 8;
        b0.q[0] = *(const v4u*)(wpg + wo); b0.q[1] = *(const v4u*)(wpg + wo + 4);
        b1.q[0] = *(const v4u*)(wpf + wo); b1.q[1] = *(const v4u*)(wpf + wo + 4);
        accg = __builtin_amdgcn_wmma_f32_16x16x32_bf16(
            false, a.v, false, b0.v, (short)0, accg, false, false);
        accf = __builtin_amdgcn_wmma_f32_16x16x32_bf16(
            false, a.v, false, b1.v, (short)0, accf, false, false);
    }

    int co = nt * 16 + m;
    float gb = bg[co], fb = bf_[co];
#pragma unroll
    for (int r = 0; r < 8; ++r) {
        float gv = accg[r] + gb;
        float fv = accf[r] + fb;
        fv = fv > 0.0f ? fv : 0.2f * fv;
        float sg = 1.0f / (1.0f + __expf(-gv));
        int xcol = xb + r + 8 * g;
        out[(((size_t)bi * 256 + co) * 128 + y) * 128 + xcol] = sg * fv;
    }
}

// =====================================================================
// host orchestration
// =====================================================================
extern "C" void kernel_launch(void* const* d_in, const int* in_sizes, int n_in,
                              void* d_out, int out_size, void* d_ws, size_t ws_size,
                              hipStream_t stream) {
    (void)in_sizes; (void)n_in; (void)out_size;

    const float* x      = (const float*)d_in[0];
    const float* xs     = (const float*)d_in[1];
    // d_in[2] = ms : mask has no effect in the reference (faithfully unused)
    const float* w_lin  = (const float*)d_in[3];
    const float* b_lin  = (const float*)d_in[4];
    const float* w_gate = (const float*)d_in[5];
    const float* b_gate = (const float*)d_in[6];
    const float* w_feat = (const float*)d_in[7];
    const float* b_feat = (const float*)d_in[8];
    float* out = (float*)d_out;

    char* base = (char*)d_ws;
    size_t off = 0;
    auto take = [&](size_t bytes) -> char* {
        char* p = base + off;
        off = (off + bytes + 255) & ~(size_t)255;
        return p;
    };
    // regions reused across the two patch sizes (stream-serialized)
    unsigned short* qbuf   = (unsigned short*)take((size_t)2 * 256 * 16384 * 2);     // 16.8 MB
    unsigned short* kbuf   = (unsigned short*)take((size_t)2 * 1280 * 16384 * 2);    // 83.9 MB
    float*          scores = (float*)         take((size_t)2 * 256 * 1280 * 4);      // 2.6 MB
    unsigned short* atten  = (unsigned short*)take((size_t)2 * 256 * 1280 * 2);      // 1.3 MB
    unsigned short* cat    = (unsigned short*)take((size_t)2 * 128 * 128 * 512 * 2); // 33.6 MB
    unsigned short* fused  = (unsigned short*)take((size_t)2 * 128 * 128 * 512 * 2); // 33.6 MB
    unsigned* wp_lin  = (unsigned*)take((size_t)589824 * 4);
    unsigned* wp_gate = (unsigned*)take((size_t)589824 * 4);
    unsigned* wp_feat = (unsigned*)take((size_t)589824 * 4);

    // optional kT region (fast GEMM2 path); deterministic given fixed ws_size
    size_t kt_bytes = (size_t)2 * 16384 * 1280 * 2;                                  // 83.9 MB
    unsigned short* ktbuf = nullptr;
    if (ws_size >= off + kt_bytes + 256) ktbuf = (unsigned short*)take(kt_bytes);

    // weight repacks + x -> fused channels 0..255 (independent of attention)
    repack_w_kernel<<<2304, 256, 0, stream>>>(w_lin,  wp_lin);
    repack_w_kernel<<<2304, 256, 0, stream>>>(w_gate, wp_gate);
    repack_w_kernel<<<2304, 256, 0, stream>>>(w_feat, wp_feat);
    x_to_fused_kernel<<<32768, 256, 0, stream>>>(x, fused);

    const int PH[2] = {8, 16}, PW[2] = {8, 16};
    for (int p = 0; p < 2; ++p) {
        int ph = PH[p], pw = PW[p];
        int oh = 128 / ph, ow = 128 / pw;
        int Q = oh * ow;
        int K = 5 * Q;
        int D = 256 * ph * pw;
        float scale = 1.0f / sqrtf((float)D);

        long long nq = 2LL * Q * D;
        pack_q_kernel<<<(unsigned)(nq / 256), 256, 0, stream>>>(x, qbuf, Q, D, ph, pw, ow);
        long long nk = 2LL * K * D;
        pack_k_kernel<<<(unsigned)(nk / 256), 256, 0, stream>>>(xs, kbuf, K, D, ph, pw, oh, ow);

        int waves1 = 2 * (Q / 16) * (K / 16);
        gemm_qk_kernel<<<waves1 / 8, 256, 0, stream>>>(qbuf, kbuf, scores, Q, K, D);

        softmax_kernel<<<2 * Q, 256, 0, stream>>>(scores, atten, K, scale);

        int waves2 = 2 * (Q / 16) * (D / 16);
        if (ktbuf) {
            pack_kt_kernel<<<(unsigned)(nk / 256), 256, 0, stream>>>(xs, ktbuf, K, D, ph, pw, oh, ow);
            gemm_av_kt_kernel<<<waves2 / 8, 256, 0, stream>>>(atten, ktbuf, cat, Q, K, D,
                                                             ph, pw, ow, p * 256);
        } else {
            gemm_av_kernel<<<waves2 / 8, 256, 0, stream>>>(atten, kbuf, cat, Q, K, D,
                                                           ph, pw, ow, p * 256);
        }
    }

    // conv_lin: cat -> fused channels 256..511 (leaky relu fused)
    conv_lin_kernel<<<4096, 256, 0, stream>>>(cat, wp_lin, b_lin, fused);

    // conv gate + feat + sigmoid*leaky fusion -> NCHW f32 output
    conv_gf_kernel<<<4096, 256, 0, stream>>>(fused, wp_gate, wp_feat, b_gate, b_feat, out);
}
